// ProximityAttention_85959475462602
// MI455X (gfx1250) — matrix operand order repaired and checked by reading, not compiled, not run on hardware
//
#include <hip/hip_runtime.h>
#include <hip/hip_bf16.h>

typedef __attribute__((ext_vector_type(16))) _Float16     v16h;
typedef __attribute__((ext_vector_type(8)))  float        v8f;
typedef __attribute__((ext_vector_type(4)))  unsigned int u32x4;
typedef __attribute__((ext_vector_type(8)))  int          i32x8;
typedef __attribute__((ext_vector_type(4)))  int          i32x4;

#define BATCH 4
#define SLEN  4096
#define DDIM  512

union FragH { v16h h; u32x4 q[2]; };

__device__ __forceinline__ v8f wmma_f16(v16h a, v16h b, v8f c) {
  // D = A(16x32 f16) * B(32x16 f16) + C(16x16 f32)
  return __builtin_amdgcn_wmma_f32_16x16x32_f16(false, a, false, b, (short)0, c,
                                                false, false);
}

// ---------------------------------------------------------------------------
// TDM: 1-D contiguous tile load global -> LDS via tensor DMA.
// D# built per CDNA5 ISA 8.3/8.4: count=1, type=2 ("image"), data_size=8B,
// tile==tensor (1-D, dims 1/2 unused), groups 2/3 zero (<=2D tensors).
// bytes must be a multiple of 8 and <= 65535*8.  6-arg builtin on this
// toolchain: (g0 v4u, g1 v8i, g2 v4i, g3 v4i, extra v8i, cpol).
// ---------------------------------------------------------------------------
__device__ __forceinline__ void tdm_load_1d(unsigned lds_off, const void* gptr,
                                            unsigned bytes) {
  const unsigned long long ga = (unsigned long long)gptr;
  const unsigned units = bytes >> 3;  // 8-byte elements
  u32x4 g0;
  g0[0] = 1u;                                                 // count=1
  g0[1] = lds_off;                                            // lds_addr (bytes)
  g0[2] = (unsigned)(ga & 0xFFFFFFFFu);                       // global_addr[31:0]
  g0[3] = (unsigned)((ga >> 32) & 0x01FFFFFFu) | (2u << 30);  // addr[56:32]|type=2
  i32x8 g1;
  g1[0] = (3 << 16);                              // data_size=3 (8B); mask/flags=0
  g1[1] = (int)((units & 0xFFFFu) << 16);         // tensor_dim0[15:0]  @bits 63:48
  g1[2] = (int)((units >> 16) & 0xFFFFu);         // tensor_dim0[31:16] @bits 79:64
  g1[3] = (int)((units & 0xFFFFu) << 16);         // tile_dim0          @bits 127:112
  g1[4] = 0;                                      // tile_dim1=0, tile_dim2=0
  g1[5] = (int)units;                             // tensor_dim0_stride[31:0]
  g1[6] = 0;
  g1[7] = 0;
  const i32x4 gz4 = {0, 0, 0, 0};
  const i32x8 gz8 = {0, 0, 0, 0, 0, 0, 0, 0};
  __builtin_amdgcn_tensor_load_to_lds(g0, g1, gz4, gz4, gz8, 0);
}

// ---------------------------------------------------------------------------
// Kernel 1: x -> f16 (row-major) + row squared norms. One wave per row.
// ---------------------------------------------------------------------------
__global__ __launch_bounds__(256) void prep_kernel(const float* __restrict__ x,
                                                   _Float16* __restrict__ xh,
                                                   float* __restrict__ sq) {
  const int wave = threadIdx.x >> 5;
  const int lane = threadIdx.x & 31;
  const size_t row = (size_t)blockIdx.x * 8 + wave;
  const float* xr = x + row * DDIM;
  _Float16* hr = xh + row * DDIM;
  float s = 0.f;
#pragma unroll
  for (int j = 0; j < DDIM / 32; ++j) {
    float v = xr[lane + 32 * j];
    s += v * v;
    hr[lane + 32 * j] = (_Float16)v;
  }
#pragma unroll
  for (int m = 1; m < 32; m <<= 1) s += __shfl_xor(s, m, 32);
  if (lane == 0) sq[row] = s;
}

// ---------------------------------------------------------------------------
// Kernel 2: W (512x512 [out,in]) -> f16.
// ---------------------------------------------------------------------------
__global__ __launch_bounds__(256) void wconv_kernel(const float* __restrict__ W,
                                                    _Float16* __restrict__ Wh) {
  const int i = blockIdx.x * 256 + threadIdx.x;
  Wh[i] = (_Float16)W[i];
}

// ---------------------------------------------------------------------------
// Kernel 3: vT[b][e][s] = f16( sum_d x[b,s,d]*W[e,d] + bias[e] )
// Batched 8-fragment B loads per k-step -> load clause + 8 back-to-back WMMAs.
// ---------------------------------------------------------------------------
__global__ __launch_bounds__(256) void vgemm_kernel(const _Float16* __restrict__ xh,
                                                    const _Float16* __restrict__ Wh,
                                                    const float* __restrict__ bias,
                                                    _Float16* __restrict__ vT) {
  const int blk  = blockIdx.x;
  const int bidx = blk / (SLEN / 32);
  const int s0   = (blk % (SLEN / 32)) * 32;
  const int wave = threadIdx.x >> 5;
  const int lane = threadIdx.x & 31;
  const int mt  = wave >> 2;
  const int eg  = wave & 3;
  const int lm  = lane & 15;
  const int lhi = lane >> 4;

  const _Float16* xb = xh + (size_t)bidx * SLEN * DDIM;

  const v8f vzero = {0.f, 0.f, 0.f, 0.f, 0.f, 0.f, 0.f, 0.f};
  v8f acc[8];
#pragma unroll
  for (int e = 0; e < 8; ++e) acc[e] = vzero;

  const _Float16* arow = xb + (size_t)(s0 + mt * 16 + lm) * DDIM;
  const _Float16* wbase = Wh + (size_t)(eg * 128 + lm) * DDIM + lhi * 16;
  for (int d0 = 0; d0 < DDIM; d0 += 32) {
    FragH a, bf[8];
    a.q[0] = *(const u32x4*)(arow + d0 + lhi * 16);
    a.q[1] = *(const u32x4*)(arow + d0 + lhi * 16 + 8);
#pragma unroll
    for (int e = 0; e < 8; ++e) {
      const _Float16* wr = wbase + e * (16 * DDIM) + d0;  // imm offsets per e
      bf[e].q[0] = ((const u32x4*)wr)[0];
      bf[e].q[1] = ((const u32x4*)wr)[1];
    }
#pragma unroll
    for (int e = 0; e < 8; ++e) acc[e] = wmma_f16(a.h, bf[e].h, acc[e]);
  }
#pragma unroll
  for (int e = 0; e < 8; ++e) {
    const int ecol = eg * 128 + e * 16 + lm;
    const float bv = bias[ecol];
    _Float16* vrow = vT + ((size_t)bidx * DDIM + ecol) * SLEN;
#pragma unroll
    for (int r = 0; r < 8; ++r) {
      const int srow = s0 + mt * 16 + 8 * lhi + r;
      vrow[srow] = (_Float16)(acc[e][r] + bv);
    }
  }
}

// ---------------------------------------------------------------------------
// Kernel 4: fused gaussian-kernel attention, TDM-pipelined.
// Dynamic-LDS layout (no statics => descriptor lds_addr == these offsets):
//   [     0, 32768)  xs    : 32 query rows x 512 f16
//   [ 32768, 36864)  wbuf  : 2 x 16 x 64 f16 probability strips
//   [ 36864, 37120)  sq_s/msk_s (32 f32 each)
//   [ 37120, 37248)  den_s (32 f32)
//   [ 40960,106496)  xt buffer 0 : 64 key rows x 512 f16 (TDM target)
//   [106496,172032)  xt buffer 1 : 64 key rows x 512 f16 (TDM target)
// Wave 0 keeps the TDM one tile ahead; s_wait_tensorcnt + barrier publish it.
// ---------------------------------------------------------------------------
#define LDS_XS    0u
#define LDS_WBUF  32768u
#define LDS_SQ    36864u
#define LDS_MSK   36992u
#define LDS_DEN   37120u
#define LDS_XT0   40960u
#define LDS_XT1   106496u
#define LDS_TOTAL 172032u

__global__ __launch_bounds__(256) void attn_kernel(const _Float16* __restrict__ xh,
                                                   const _Float16* __restrict__ vT,
                                                   const float* __restrict__ sq,
                                                   const unsigned char* __restrict__ mask,
                                                   float* __restrict__ out) {
  extern __shared__ char smem[];
  _Float16* xs   = (_Float16*)(smem + LDS_XS);
  _Float16* wbuf = (_Float16*)(smem + LDS_WBUF);
  float* sq_s  = (float*)(smem + LDS_SQ);
  float* msk_s = (float*)(smem + LDS_MSK);
  float* den_s = (float*)(smem + LDS_DEN);

  const int blk  = blockIdx.x;
  const int bidx = blk / (SLEN / 32);
  const int row0 = (blk % (SLEN / 32)) * 32;
  const int tid  = threadIdx.x;
  const int wave = tid >> 5;
  const int lane = tid & 31;
  const int mt  = wave >> 2;
  const int tg  = wave & 3;
  const int lm  = lane & 15;
  const int lhi = lane >> 4;

  const _Float16* xb  = xh + (size_t)bidx * SLEN * DDIM;
  const _Float16* vTb = vT + (size_t)bidx * DDIM * SLEN;
  const float* sqb = sq + (size_t)bidx * SLEN;
  const unsigned char* mb = mask + (size_t)bidx * SLEN;

  {  // cooperative 32 KB copy: 32 contiguous query rows -> LDS
    const u32x4* src = (const u32x4*)(xb + (size_t)row0 * DDIM);
    u32x4* dst = (u32x4*)xs;
    for (int i = tid; i < 32 * DDIM / 8; i += 256) dst[i] = src[i];
  }
  if (tid < 32) {
    sq_s[tid]  = sqb[row0 + tid];
    msk_s[tid] = mb[row0 + tid] ? 1.f : 0.f;
    den_s[tid] = 0.f;
  }
  if (wave == 0) {  // TDM: key tile 0 -> xt buffer 0 (64 rows x 1 KB = 64 KB)
    tdm_load_1d(LDS_XT0, xb, 64 * DDIM * 2);
  }
  __syncthreads();

  float sqm[8];
#pragma unroll
  for (int r = 0; r < 8; ++r) sqm[r] = sq_s[mt * 16 + 8 * lhi + r];

  const v8f vzero = {0.f, 0.f, 0.f, 0.f, 0.f, 0.f, 0.f, 0.f};
  v8f acc[8];
#pragma unroll
  for (int e = 0; e < 8; ++e) acc[e] = vzero;
  float dsum[8] = {0.f, 0.f, 0.f, 0.f, 0.f, 0.f, 0.f, 0.f};

  const _Float16* as_row = xs + (size_t)(mt * 16 + lm) * DDIM;
  const _Float16* wb_row = wbuf + (size_t)(mt * 16 + lm) * 64;
  // loop-invariant per-lane vT base: per-e / per-j deltas are compile-time
  // immediates (e stride 16*SLEN halfs = 128 KB < 24-bit imm range).
  const _Float16* vlane = vTb + (size_t)(tg * 128 + lm) * SLEN + lhi * 16;

  for (int t0 = 0; t0 < SLEN; t0 += 64) {
    const int cur = (t0 >> 6) & 1;
    // ---- TDM pipeline: issue tile i+1, wait for tile i, publish ------------
    if (wave == 0) {
      if (t0 + 64 < SLEN) {
        tdm_load_1d(cur ? LDS_XT0 : LDS_XT1,
                    xb + (size_t)(t0 + 64) * DDIM, 64 * DDIM * 2);
        __builtin_amdgcn_s_wait_tensorcnt((short)1);
      } else {
        __builtin_amdgcn_s_wait_tensorcnt((short)0);
      }
    }
    __syncthreads();

    const int tcol = t0 + tg * 16;
    // current key-tile buffer: integer-offset select, no pointer-array init
    const _Float16* xtc = (const _Float16*)(smem + (cur ? LDS_XT1 : LDS_XT0));
    // ---- score tile: dot(x_m, x_t) over D=512, all operands in LDS --------
    v8f sc = vzero;
    const _Float16* brow = xtc + (size_t)(tg * 16 + lm) * DDIM;
    for (int d0 = 0; d0 < DDIM; d0 += 32) {
      FragH a, bf;
      a.q[0] = *(const u32x4*)(as_row + d0 + lhi * 16);
      a.q[1] = *(const u32x4*)(as_row + d0 + lhi * 16 + 8);
      bf.q[0] = ((const u32x4*)(brow + d0 + lhi * 16))[0];
      bf.q[1] = ((const u32x4*)(brow + d0 + lhi * 16))[1];
      sc = wmma_f16(a.h, bf.h, sc);
    }
    // ---- gaussian weight, masked; w <= 1 so f16 is safe --------------------
    const float sqt  = sqb[tcol + lm];
    const float mskt = mb[tcol + lm] ? 1.f : 0.f;
    _Float16 wh[8];
#pragma unroll
    for (int r = 0; r < 8; ++r) {
      float d2 = sqm[r] + sqt - 2.f * sc[r];
      d2 = fmaxf(d2, 0.f);
      float w = __expf(-d2) * mskt;
      dsum[r] += w;
      wh[r] = (_Float16)w;
    }
    __syncthreads();  // previous iteration's wbuf readers are done
#pragma unroll
    for (int r = 0; r < 8; ++r)
      wbuf[(mt * 16 + 8 * lhi + r) * 64 + tg * 16 + lm] = wh[r];
    __syncthreads();
    // ---- PV: out[m, e-slice] += w[m, t-tile] @ v[t-tile, e-slice] ----------
    const _Float16* vt0 = vlane + t0;  // one pointer bump per iteration
#pragma unroll
    for (int j = 0; j < 2; ++j) {
      FragH a, bf[8];
      a.q[0] = *(const u32x4*)(wb_row + j * 32 + lhi * 16);
      a.q[1] = *(const u32x4*)(wb_row + j * 32 + lhi * 16 + 8);
#pragma unroll
      for (int e = 0; e < 8; ++e) {
        const _Float16* vr = vt0 + e * (16 * SLEN) + j * 32;  // imm offsets
        bf[e].q[0] = ((const u32x4*)vr)[0];
        bf[e].q[1] = ((const u32x4*)vr)[1];
      }
#pragma unroll
      for (int e = 0; e < 8; ++e) acc[e] = wmma_f16(a.h, bf[e].h, acc[e]);
    }
    if (t0 + 64 < SLEN) {  // pull next vT tile toward L2/L0
      __builtin_prefetch(vt0 + 64, 0, 0);
    }
  }

  // ---- denominator: column reduce in-wave, combine across the 4 tg waves ---
#pragma unroll
  for (int m = 1; m < 16; m <<= 1) {
#pragma unroll
    for (int r = 0; r < 8; ++r) dsum[r] += __shfl_xor(dsum[r], m, 32);
  }
  if (lm == 0) {
#pragma unroll
    for (int r = 0; r < 8; ++r)
      atomicAdd(&den_s[mt * 16 + 8 * lhi + r], dsum[r]);
  }
  __syncthreads();

  // ---- epilogue: out = mask_s * num / max(denom, eps) ----------------------
#pragma unroll
  for (int r = 0; r < 8; ++r) {
    const int m = mt * 16 + 8 * lhi + r;
    const float scale = msk_s[m] / fmaxf(den_s[m], 1e-8f);
    float* orow = out + ((size_t)bidx * SLEN + (row0 + m)) * DDIM + tg * 128 + lm;
#pragma unroll
    for (int e = 0; e < 8; ++e) orow[e * 16] = acc[e][r] * scale;
  }
}

// ---------------------------------------------------------------------------
extern "C" void kernel_launch(void* const* d_in, const int* in_sizes, int n_in,
                              void* d_out, int out_size, void* d_ws, size_t ws_size,
                              hipStream_t stream) {
  (void)in_sizes; (void)n_in; (void)out_size; (void)ws_size;
  const float* x            = (const float*)d_in[0];
  const unsigned char* mask = (const unsigned char*)d_in[1];  // jax bool = 1 byte
  const float* W            = (const float*)d_in[2];
  const float* bias         = (const float*)d_in[3];
  float* out                = (float*)d_out;

  // workspace layout (all regions fully rewritten each call): ~33 MB
  char* ws = (char*)d_ws;
  const size_t off_xh = 0;                                          // 16 MB
  const size_t off_wh = off_xh + (size_t)BATCH * SLEN * DDIM * 2;   // +0.5 MB
  const size_t off_sq = off_wh + (size_t)DDIM * DDIM * 2;           // +64 KB
  const size_t off_vt = off_sq + (size_t)BATCH * SLEN * 4;          // +16 MB
  _Float16* xh = (_Float16*)(ws + off_xh);
  _Float16* Wh = (_Float16*)(ws + off_wh);
  float*    sq = (float*)(ws + off_sq);
  _Float16* vT = (_Float16*)(ws + off_vt);

  prep_kernel <<<(BATCH * SLEN) / 8,  256, 0, stream>>>(x, xh, sq);
  wconv_kernel<<<(DDIM * DDIM) / 256, 256, 0, stream>>>(W, Wh);
  vgemm_kernel<<<BATCH * (SLEN / 32), 256, 0, stream>>>(xh, Wh, bias, vT);
  attn_kernel <<<BATCH * (SLEN / 32), 256, LDS_TOTAL, stream>>>(xh, vT, sq, mask, out);
}
